// TransformerBlock_50551765074595
// MI455X (gfx1250) — matrix operand order repaired and checked by reading, not compiled, not run on hardware
//
#include <hip/hip_runtime.h>
#include <hip/hip_bf16.h>

typedef __attribute__((ext_vector_type(16))) __bf16 v16bf;
typedef __attribute__((ext_vector_type(8)))  __bf16 v8bf;
typedef __attribute__((ext_vector_type(8)))  float  v8f;

static constexpr int EMBED  = 1024;
static constexpr int HIDDEN = 4096;
static constexpr int SEQ    = 2048;
static constexpr int BATCH  = 4;
static constexpr int ROWS   = BATCH * SEQ;   // 8192

#define MODE_RELU  1
#define MODE_TRANS 2
#define MODE_F32   4
#define MODE_RES   8

#define BM 128
#define BN 128
#define BK 32
#define LDSA 40   // padded LDS row stride (halfs): 80B, keeps 16B alignment
#define LDSB 40

// gfx1250 async global->LDS path (ASYNCcnt-tracked, bypasses VGPRs).
// Guarded so host pass / older toolchains fall back to sync staging.
#if defined(__AMDGCN__) && __has_builtin(__builtin_amdgcn_global_load_async_to_lds_b128) && __has_builtin(__builtin_amdgcn_s_wait_asynccnt)
#define HAVE_ASYNC_LDS 1
#else
#define HAVE_ASYNC_LDS 0
#endif

#if HAVE_ASYNC_LDS
typedef int v4i __attribute__((vector_size(16)));
typedef __attribute__((address_space(3))) v4i* as3_v4i;
__device__ __forceinline__ void async_copy16(const void* g, void* l) {
  // generic->AS3 via 32-bit truncation (flat->local addrspacecast = low 32 bits)
  __builtin_amdgcn_global_load_async_to_lds_b128(
      (v4i*)g, (as3_v4i)(unsigned int)(unsigned long long)l, 0, 0);
}
#endif

// ---- WMMA fragment loaders (layouts per CDNA5 ISA 7.12.2) ----
// A (16x32 bf16): lane holds row M=lane&15; halfs [0..7]=K h*8.., [8..15]=K 16+h*8..
__device__ __forceinline__ v16bf frag_a(const __bf16* row, int h) {
  v8bf lo = *(const v8bf*)(row + h * 8);
  v8bf hi = *(const v8bf*)(row + 16 + h * 8);
  return __builtin_shufflevector(lo, hi, 0,1,2,3,4,5,6,7,8,9,10,11,12,13,14,15);
}
// B (32x16 bf16): lane holds col N=lane&15; halfs [0..7]=K h*8.., [8..15]=K 16+h*8..
__device__ __forceinline__ v16bf frag_b(const __bf16* row, int h) {
  v8bf lo = *(const v8bf*)(row + h * 8);
  v8bf hi = *(const v8bf*)(row + 16 + h * 8);
  return __builtin_shufflevector(lo, hi, 0,1,2,3,4,5,6,7,8,9,10,11,12,13,14,15);
}

// ---- Generic bf16 GEMM: C = epilogue( A[M,K] @ Bt[N,K]^T ) ----
// Block tile 128x128, 8 waves, each wave a 32x64 tile (2x4 WMMA accumulators).
__global__ __launch_bounds__(256) void gemm_bf16_kernel(
    const __bf16* __restrict__ A, int lda, long long sA,
    const __bf16* __restrict__ B, int ldb, long long sB,
    const float* __restrict__ bias, float scale, int mode,
    void* __restrict__ C, int ldc, long long sC,
    const float* __restrict__ Res, long long sRes,
    int K)
{
#if HAVE_ASYNC_LDS
  __shared__ __align__(16) __bf16 As[2][BM * LDSA];
  __shared__ __align__(16) __bf16 Bs[2][BN * LDSB];
#else
  __shared__ __align__(16) __bf16 As[1][BM * LDSA];
  __shared__ __align__(16) __bf16 Bs[1][BN * LDSB];
#endif

  const int t    = threadIdx.x;
  const int lane = t & 31;
  const int wave = t >> 5;
  const int wm   = wave & 3;   // 4 waves along M (32 rows each)
  const int wn   = wave >> 2;  // 2 waves along N (64 cols each)
  const int lr   = lane & 15;
  const int h    = lane >> 4;
  const int bz   = blockIdx.z;

  const __bf16* Ab = A + (long long)bz * sA;
  const __bf16* Bb = B + (long long)bz * sB;
  const long long rowBase = (long long)blockIdx.y * BM;
  const long long colBase = (long long)blockIdx.x * BN;

  v8f zero = {0.f,0.f,0.f,0.f,0.f,0.f,0.f,0.f};
  v8f acc[2][4];
  #pragma unroll
  for (int mi = 0; mi < 2; ++mi)
    #pragma unroll
    for (int ni = 0; ni < 4; ++ni) acc[mi][ni] = zero;

  const int nk = K / BK;

#if HAVE_ASYNC_LDS
  // Stage one 128x32 A tile + 128x32 B tile: 512+512 16B chunks, 4 per thread.
  auto issue_tile = [&](int buf, int k0) {
    #pragma unroll
    for (int i = 0; i < 2; ++i) {
      int cid = t + i * 256;
      int r = cid >> 2, c = cid & 3;
      async_copy16(Ab + (rowBase + r) * lda + k0 + c * 8, &As[buf][r * LDSA + c * 8]);
      async_copy16(Bb + (colBase + r) * ldb + k0 + c * 8, &Bs[buf][r * LDSB + c * 8]);
    }
  };
  issue_tile(0, 0);
#endif

  for (int i = 0; i < nk; ++i) {
#if HAVE_ASYNC_LDS
    const int cur = i & 1;
    __builtin_amdgcn_s_wait_asynccnt(0); // current tile resident (next not yet issued)
    __syncthreads();                     // all waves' tile-i copies done; prior reads of buf[i^1] done
    if (i + 1 < nk) issue_tile(cur ^ 1, (i + 1) * BK); // overlaps with WMMA below
#else
    const int cur = 0;
    const int k0 = i * BK;
    #pragma unroll
    for (int j = 0; j < 2; ++j) {
      int cid = t + j * 256;
      int r = cid >> 2, c = cid & 3;
      *(uint4*)(&As[0][r * LDSA + c * 8]) = *(const uint4*)(Ab + (rowBase + r) * lda + k0 + c * 8);
      *(uint4*)(&Bs[0][r * LDSB + c * 8]) = *(const uint4*)(Bb + (colBase + r) * ldb + k0 + c * 8);
    }
    if (i + 1 < nk) {
      __builtin_prefetch(Ab + (rowBase + (t >> 2)) * lda + k0 + BK, 0, 3);
      __builtin_prefetch(Bb + (colBase + (t >> 2)) * ldb + k0 + BK, 0, 3);
    }
    __syncthreads();
#endif

    v16bf af[2], bfr[4];
    #pragma unroll
    for (int mi = 0; mi < 2; ++mi)
      af[mi] = frag_a(&As[cur][(wm * 32 + mi * 16 + lr) * LDSA], h);
    #pragma unroll
    for (int ni = 0; ni < 4; ++ni)
      bfr[ni] = frag_b(&Bs[cur][(wn * 64 + ni * 16 + lr) * LDSB], h);

    #pragma unroll
    for (int mi = 0; mi < 2; ++mi)
      #pragma unroll
      for (int ni = 0; ni < 4; ++ni)
        acc[mi][ni] = __builtin_amdgcn_wmma_f32_16x16x32_bf16(
            false, af[mi], false, bfr[ni], (short)0, acc[mi][ni], false, false);

#if !HAVE_ASYNC_LDS
    __syncthreads();
#endif
  }

  // Epilogue. C/D layout: lane holds N=lr, M=(h*8+v) within each 16x16 tile.
  #pragma unroll
  for (int mi = 0; mi < 2; ++mi) {
    #pragma unroll
    for (int ni = 0; ni < 4; ++ni) {
      long long gn  = colBase + wn * 64 + ni * 16 + lr;
      long long gm0 = rowBase + wm * 32 + mi * 16 + h * 8;
      float bval = bias ? bias[gn] : 0.0f;
      #pragma unroll
      for (int v = 0; v < 8; ++v) {
        long long m = gm0 + v;
        float cv = (acc[mi][ni][v] + bval) * scale;
        if (mode & MODE_RELU) cv = fmaxf(cv, 0.0f);
        if (mode & MODE_RES)  cv += Res[(long long)bz * sRes + m * ldc + gn];
        if (mode & MODE_F32) {
          ((float*)C)[(long long)bz * sC + m * ldc + gn] = cv;
        } else if (mode & MODE_TRANS) {
          // batch-transposed bf16 out: Vt[b][d][s], b=m>>11, s=m&2047
          ((__bf16*)C)[((m >> 11) * (long long)EMBED + gn) * SEQ + (m & 2047)] = (__bf16)cv;
        } else {
          ((__bf16*)C)[(long long)bz * sC + m * ldc + gn] = (__bf16)cv;
        }
      }
    }
  }
}

// ---- LayerNorm: one block per row of 1024, f32 in -> bf16 out ----
__global__ __launch_bounds__(256) void ln_kernel(
    const float* __restrict__ x, const float* __restrict__ g,
    const float* __restrict__ b, __bf16* __restrict__ out)
{
  __shared__ float red[256];
  const int t = threadIdx.x;
  const long long row = blockIdx.x;
  const float* xr = x + row * EMBED;

  float v[4]; float s = 0.0f;
  #pragma unroll
  for (int i = 0; i < 4; ++i) { v[i] = xr[t + i * 256]; s += v[i]; }
  red[t] = s; __syncthreads();
  for (int o = 128; o > 0; o >>= 1) { if (t < o) red[t] += red[t + o]; __syncthreads(); }
  float mu = red[0] * (1.0f / EMBED);
  __syncthreads();

  float sv = 0.0f;
  #pragma unroll
  for (int i = 0; i < 4; ++i) { float d = v[i] - mu; sv += d * d; }
  red[t] = sv; __syncthreads();
  for (int o = 128; o > 0; o >>= 1) { if (t < o) red[t] += red[t + o]; __syncthreads(); }
  float inv = rsqrtf(red[0] * (1.0f / EMBED) + 1e-5f);

  #pragma unroll
  for (int i = 0; i < 4; ++i) {
    int idx = t + i * 256;
    out[row * EMBED + idx] = (__bf16)((v[i] - mu) * inv * g[idx] + b[idx]);
  }
}

// ---- Softmax over rows of 2048 f32 scores; writes bf16 probs in place ----
__global__ __launch_bounds__(256) void softmax_kernel(float* __restrict__ S)
{
  __shared__ float red[256];
  const int t = threadIdx.x;
  float* row = S + (long long)blockIdx.x * SEQ;

  float v[8]; float mx = -1e30f;
  #pragma unroll
  for (int i = 0; i < 8; ++i) { v[i] = row[t + i * 256]; mx = fmaxf(mx, v[i]); }
  red[t] = mx; __syncthreads();
  for (int o = 128; o > 0; o >>= 1) { if (t < o) red[t] = fmaxf(red[t], red[t + o]); __syncthreads(); }
  mx = red[0]; __syncthreads();

  float s = 0.0f;
  #pragma unroll
  for (int i = 0; i < 8; ++i) { v[i] = __expf(v[i] - mx); s += v[i]; }
  red[t] = s; __syncthreads();
  for (int o = 128; o > 0; o >>= 1) { if (t < o) red[t] += red[t + o]; __syncthreads(); }
  float inv = 1.0f / red[0];
  __syncthreads();   // all loads done before in-place bf16 overwrite

  __bf16* o = (__bf16*)row;   // row-local overlay; probs packed at row start
  #pragma unroll
  for (int i = 0; i < 8; ++i) o[t + i * 256] = (__bf16)(v[i] * inv);
}

// ---- Weight convert+transpose: W[K][N] f32 -> Wt[N][K] bf16 ----
__global__ __launch_bounds__(256) void wtrans_kernel(
    const float* __restrict__ W, __bf16* __restrict__ Wt, int K, int N)
{
  long long idx = (long long)blockIdx.x * 256 + threadIdx.x;
  int n = (int)(idx / K);
  int k = (int)(idx - (long long)n * K);
  Wt[idx] = (__bf16)W[(long long)k * N + n];
}

extern "C" void kernel_launch(void* const* d_in, const int* in_sizes, int n_in,
                              void* d_out, int out_size, void* d_ws, size_t ws_size,
                              hipStream_t stream) {
  (void)in_sizes; (void)n_in; (void)out_size; (void)ws_size;
  const float* x   = (const float*)d_in[0];
  const float* Wq  = (const float*)d_in[1];
  const float* bq  = (const float*)d_in[2];
  const float* Wk  = (const float*)d_in[3];
  const float* bk  = (const float*)d_in[4];
  const float* Wv  = (const float*)d_in[5];
  const float* bv  = (const float*)d_in[6];
  const float* W1  = (const float*)d_in[7];
  const float* b1  = (const float*)d_in[8];
  const float* W2  = (const float*)d_in[9];
  const float* b2  = (const float*)d_in[10];
  const float* g1  = (const float*)d_in[11];
  const float* be1 = (const float*)d_in[12];
  const float* g2  = (const float*)d_in[13];
  const float* be2 = (const float*)d_in[14];
  float* out = (float*)d_out;

  char* ws = (char*)d_ws;
  size_t off = 0;
  auto alloc = [&](size_t bytes) {
    char* p = ws + off;
    off = (off + bytes + 255) & ~(size_t)255;
    return p;
  };
  __bf16* WqT = (__bf16*)alloc((size_t)EMBED * EMBED * 2);       //  2 MB
  __bf16* WkT = (__bf16*)alloc((size_t)EMBED * EMBED * 2);       //  2 MB
  __bf16* WvT = (__bf16*)alloc((size_t)EMBED * EMBED * 2);       //  2 MB
  __bf16* W1T = (__bf16*)alloc((size_t)HIDDEN * EMBED * 2);      //  8 MB
  __bf16* W2T = (__bf16*)alloc((size_t)EMBED * HIDDEN * 2);      //  8 MB
  __bf16* hbf = (__bf16*)alloc((size_t)ROWS * EMBED * 2);        // 16 MB (LN1, reused LN2)
  __bf16* Qb  = (__bf16*)alloc((size_t)ROWS * EMBED * 2);        // 16 MB
  __bf16* Kb  = (__bf16*)alloc((size_t)ROWS * EMBED * 2);        // 16 MB
  __bf16* Vt  = (__bf16*)alloc((size_t)BATCH * EMBED * SEQ * 2); // 16 MB
  float*  big = (float*)alloc((size_t)BATCH * SEQ * SEQ * 4);    // 64 MB (scores/P, then ff1)
  float*  x1  = (float*)alloc((size_t)ROWS * EMBED * 4);         // 32 MB

  dim3 blk(256);

  // 1) bf16-transpose all weights
  wtrans_kernel<<<dim3(EMBED * EMBED / 256), blk, 0, stream>>>(Wq, WqT, EMBED, EMBED);
  wtrans_kernel<<<dim3(EMBED * EMBED / 256), blk, 0, stream>>>(Wk, WkT, EMBED, EMBED);
  wtrans_kernel<<<dim3(EMBED * EMBED / 256), blk, 0, stream>>>(Wv, WvT, EMBED, EMBED);
  wtrans_kernel<<<dim3(EMBED * HIDDEN / 256), blk, 0, stream>>>(W1, W1T, EMBED, HIDDEN);
  wtrans_kernel<<<dim3(HIDDEN * EMBED / 256), blk, 0, stream>>>(W2, W2T, HIDDEN, EMBED);

  // 2) h = LN1(x)
  ln_kernel<<<dim3(ROWS), blk, 0, stream>>>(x, g1, be1, hbf);

  // 3) Q/K projections (bf16 out), V projection stored batch-transposed
  dim3 gproj(EMBED / BN, ROWS / BM, 1);
  gemm_bf16_kernel<<<gproj, blk, 0, stream>>>(hbf, EMBED, 0, WqT, EMBED, 0,
      bq, 1.0f, 0, Qb, EMBED, 0, nullptr, 0, EMBED);
  gemm_bf16_kernel<<<gproj, blk, 0, stream>>>(hbf, EMBED, 0, WkT, EMBED, 0,
      bk, 1.0f, 0, Kb, EMBED, 0, nullptr, 0, EMBED);
  gemm_bf16_kernel<<<gproj, blk, 0, stream>>>(hbf, EMBED, 0, WvT, EMBED, 0,
      bv, 1.0f, MODE_TRANS, Vt, 0, 0, nullptr, 0, EMBED);

  // 4) scores = (Q @ K^T) / sqrt(D), batched over z, f32 out
  dim3 gsc(SEQ / BN, SEQ / BM, BATCH);
  gemm_bf16_kernel<<<gsc, blk, 0, stream>>>(
      Qb, EMBED, (long long)SEQ * EMBED, Kb, EMBED, (long long)SEQ * EMBED,
      nullptr, 0.03125f, MODE_F32, big, SEQ, (long long)SEQ * SEQ,
      nullptr, 0, EMBED);

  // 5) softmax rows -> bf16 probs in place
  softmax_kernel<<<dim3(ROWS), blk, 0, stream>>>(big);

  // 6) x1 = x + P @ V   (A = bf16 probs with half-stride 4096 over the f32 rows)
  dim3 gat(EMBED / BN, SEQ / BM, BATCH);
  gemm_bf16_kernel<<<gat, blk, 0, stream>>>(
      (const __bf16*)big, 2 * SEQ, (long long)SEQ * 2 * SEQ,
      Vt, SEQ, (long long)EMBED * SEQ,
      nullptr, 1.0f, MODE_F32 | MODE_RES, x1, EMBED, (long long)SEQ * EMBED,
      x, (long long)SEQ * EMBED, SEQ);

  // 7) h2 = LN2(x1)
  ln_kernel<<<dim3(ROWS), blk, 0, stream>>>(x1, g2, be2, hbf);

  // 8) ff1 = relu(h2 @ W1 + b1), bf16 into big
  dim3 gff1(HIDDEN / BN, ROWS / BM, 1);
  gemm_bf16_kernel<<<gff1, blk, 0, stream>>>(hbf, EMBED, 0, W1T, EMBED, 0,
      b1, 1.0f, MODE_RELU, big, HIDDEN, 0, nullptr, 0, EMBED);

  // 9) out = x1 + ff1 @ W2 + b2
  dim3 gff2(EMBED / BN, ROWS / BM, 1);
  gemm_bf16_kernel<<<gff2, blk, 0, stream>>>(
      (const __bf16*)big, HIDDEN, 0, W2T, HIDDEN, 0,
      b2, 1.0f, MODE_F32 | MODE_RES, out, EMBED, 0, x1, 0, HIDDEN);
}
